// UpdateNode_24927990186016
// MI455X (gfx1250) — matrix operand order repaired and checked by reading, not compiled, not run on hardware
//
#include <hip/hip_runtime.h>
#include <hip/hip_bf16.h>

typedef __bf16 bf16;
typedef __attribute__((ext_vector_type(16))) __bf16 v16bf;
typedef __attribute__((ext_vector_type(8)))  float  v8f;

#define E_TOT   160000
#define NNODES  25000
#define ACC_N   (25000 * 240)

// bf16 weight arena offsets (elements) inside d_ws after the fp32 accumulator.
// Each matrix is stored FRAGMENT-SWIZZLED: tiles of 32(K)x16(N), tile index
// t = (n0/16)*KT + (k0/32), and inside a tile: lane L (0..31) holds 16
// contiguous bf16 = the exact V_WMMA_16X16X32 B-operand fragment.
#define OFF_WLAT 0         // 128x224
#define OFF_WM0  28672     // 224x160
#define OFF_WM1C 64512     // 192x96  [[Wm1r, Wm1i],[-Wm1i, Wm1r]]
#define OFF_WM2C 82944     // 64x32   [[Wm2r, Wm2i],[-Wm2i, Wm2r]]
#define OFF_WENV 84992     // 128x112
#define OFF_WP0  99328     // 64x64
#define OFF_WR0  103424    // 64x64
#define OFF_WR1  107520    // 32x32
#define OFF_WR2P 108544    // 32x16 (K padded 16->32 with zeros)
#define OFF_WOC  109056    // 96x112 (one-hot table, row 95 zero pad; [Wo0|Wo1|Wo2])
#define NW_TOTAL 119808

#define LANE (threadIdx.x & 31)

__device__ __forceinline__ float sigm(float x) { return 1.f / (1.f + __expf(-x)); }

// ---- WMMA fragment helpers (16x16x32 bf16, wave32) -------------------------
// A (16x32 16-bit): lane half h holds K-blocks {8h..8h+7, 16+8h..16+8h+7}
__device__ __forceinline__ v16bf frag_A_lds(const bf16* A, int lda, int k0) {
  int m = LANE & 15, half = LANE >> 4;
  v16bf a;
#pragma unroll
  for (int e = 0; e < 16; ++e) {
    int v = e >> 1, p = e & 1;
    int k = (v < 4) ? (half * 8 + 2 * v + p) : (16 + half * 8 + 2 * (v - 4) + p);
    a[e] = A[m * lda + k0 + k];
  }
  return a;
}
// B fragment from the swizzled arena: one aligned 32-byte vector load per lane
__device__ __forceinline__ v16bf frag_B_sw(const bf16* Bs, int kt, int k0, int n0) {
  int tile = (n0 >> 4) * kt + (k0 >> 5);
  return *(const v16bf*)(Bs + (tile << 9) + (LANE << 4));
}
// D (16x16 f32): VGPR r, lane L -> row = r + 8*(L/16), col = L%16
__device__ __forceinline__ void store_D(float* C, int ldc, int n0, v8f d) {
  int col = LANE & 15, half = LANE >> 4;
#pragma unroll
  for (int r = 0; r < 8; ++r) C[(r + 8 * half) * ldc + n0 + col] = d[r];
}

// GEMM with A fragments hoisted into registers (KT = K/32 tiles)
template <int KT>
__device__ void gemm_bf(const bf16* A, int lda, const bf16* Bs,
                        float* C, int ldc, int N) {
  v16bf a[KT];
#pragma unroll
  for (int kk = 0; kk < KT; ++kk) a[kk] = frag_A_lds(A, lda, kk * 32);
  for (int n0 = 0; n0 < N; n0 += 16) {
    v8f acc = {};
#pragma unroll
    for (int kk = 0; kk < KT; ++kk) {
      v16bf b = frag_B_sw(Bs, KT, kk * 32, n0);
      acc = __builtin_amdgcn_wmma_f32_16x16x32_bf16(false, a[kk], false, b,
                                                    (short)0, acc, false, false);
    }
    store_D(C, ldc, n0, acc);
  }
}

// A sourced from strided fp32 LDS, converted on the fly, zero-padded past klim
template <int KT>
__device__ void gemm_f32A(const float* A, int rstride, int cbase, int cstride,
                          int klim, const bf16* Bs, float* C, int ldc, int N) {
  int m = LANE & 15, half = LANE >> 4;
  v16bf a[KT];
#pragma unroll
  for (int kk = 0; kk < KT; ++kk) {
#pragma unroll
    for (int e = 0; e < 16; ++e) {
      int v = e >> 1, p = e & 1;
      int k = (v < 4) ? (half * 8 + 2 * v + p) : (16 + half * 8 + 2 * (v - 4) + p);
      int kkk = kk * 32 + k;
      float x = (kkk < klim) ? A[m * rstride + cbase + kkk * cstride] : 0.f;
      a[kk][e] = (bf16)x;
    }
  }
  for (int n0 = 0; n0 < N; n0 += 16) {
    v8f acc = {};
#pragma unroll
    for (int kk = 0; kk < KT; ++kk) {
      v16bf b = frag_B_sw(Bs, KT, kk * 32, n0);
      acc = __builtin_amdgcn_wmma_f32_16x16x32_bf16(false, a[kk], false, b,
                                                    (short)0, acc, false, false);
    }
    store_D(C, ldc, n0, acc);
  }
}

// decode swizzled element index j (within one matrix) -> (k, n)
__device__ __forceinline__ void dec_sw(int j, int K, int N, int& k, int& n) {
  int kt = K >> 5;
  int t = j >> 9, r = j & 511;
  int L = r >> 4, e = r & 15;
  int ktid = t % kt, ntid = t / kt;
  k = ktid * 32 + ((e >> 3) << 4) + ((L >> 4) << 3) + (e & 7);
  n = ntid * 16 + (L & 15);
  (void)N;
}

// ---- setup: zero accumulator, build swizzled bf16 weight arena -------------
__global__ void setup_kernel(float* acc, bf16* WB,
    const float* Wlat, const float* Wm0, const float* Wm1r, const float* Wm1i,
    const float* Wm2r, const float* Wm2i, const float* Wenv, const float* Wp0,
    const float* Wr0, const float* Wr1, const float* Wr2,
    const float* Wo0, const float* Wo1, const float* Wo2) {
  int stride = gridDim.x * blockDim.x;
  int tid = blockIdx.x * blockDim.x + threadIdx.x;
  for (int i = tid; i < ACC_N; i += stride) acc[i] = 0.f;
  for (int i = tid; i < NW_TOTAL; i += stride) {
    float v; int k, n;
    if (i < OFF_WM0) { dec_sw(i, 128, 224, k, n); v = Wlat[k * 224 + n]; }
    else if (i < OFF_WM1C) { dec_sw(i - OFF_WM0, 224, 160, k, n); v = Wm0[k * 160 + n]; }
    else if (i < OFF_WM2C) {
      dec_sw(i - OFF_WM1C, 192, 96, k, n);
      if (n < 48) v = (k < 96) ? Wm1r[k * 48 + n] : -Wm1i[(k - 96) * 48 + n];
      else        v = (k < 96) ? Wm1i[k * 48 + n - 48] : Wm1r[(k - 96) * 48 + n - 48];
    } else if (i < OFF_WENV) {
      dec_sw(i - OFF_WM2C, 64, 32, k, n);
      if (n < 16) v = (k < 32) ? Wm2r[k * 16 + n] : -Wm2i[(k - 32) * 16 + n];
      else        v = (k < 32) ? Wm2i[k * 16 + n - 16] : Wm2r[(k - 32) * 16 + n - 16];
    }
    else if (i < OFF_WP0)  { dec_sw(i - OFF_WENV, 128, 112, k, n); v = Wenv[k * 112 + n]; }
    else if (i < OFF_WR0)  { dec_sw(i - OFF_WP0,   64,  64, k, n); v = Wp0[k * 64 + n]; }
    else if (i < OFF_WR1)  { dec_sw(i - OFF_WR0,   64,  64, k, n); v = Wr0[k * 64 + n]; }
    else if (i < OFF_WR2P) { dec_sw(i - OFF_WR1,   32,  32, k, n); v = Wr1[k * 32 + n]; }
    else if (i < OFF_WOC)  {
      dec_sw(i - OFF_WR2P, 32, 16, k, n);
      v = (k < 16) ? Wr2[k * 16 + n] : 0.f;
    } else {
      dec_sw(i - OFF_WOC, 96, 112, k, n);
      if (k >= 95)     v = 0.f;
      else if (n < 64) v = Wo0[k * 64 + n];
      else if (n < 96) v = Wo1[k * 32 + (n - 64)];
      else             v = Wo2[k * 16 + (n - 96)];
    }
    WB[i] = (bf16)v;
  }
}

// ---- edge kernel: one wave = 16 edges --------------------------------------
__global__ __launch_bounds__(32) void edge_kernel(
    const float* __restrict__ latents, const float* __restrict__ node_feat,
    const float* __restrict__ edge_feat, const int* __restrict__ edge_index,
    const int* __restrict__ active_edges, const float* __restrict__ wigner,
    const float* __restrict__ Wp1, const float* __restrict__ Wp2,
    const float* __restrict__ bp0, const bf16* __restrict__ WB,
    float* __restrict__ acc) {
  __shared__ char smem[59392];
  bf16*  lat_bf  = (bf16*) (smem +     0);   // 16x128 bf16
  float* wbuf    = (float*)(smem +  4096);   // 16x224 f32 (dead after phase B)
  float* webuf   = (float*)(smem + 18432);   // 16x112 f32 (live to end)
  bf16*  in0_bf  = (bf16*) (smem + 25600);   // 16x224 bf16
  bf16*  xpn_bf  = (bf16*) (smem + 32768);   // 16x192 bf16
  bf16*  m2_bf   = (bf16*) (smem + 38912);   // 16x64  bf16
  float* out0    = (float*)(smem + 40960);   // 16x160 f32
  float* ypn     = (float*)(smem + 51200);   // 16x96  f32
  float* y2c     = (float*)(smem + 57344);   // 16x32  f32
  float* y1b     = (float*)(smem +  4096);   // overlay: 16x96 f32
  float* y2b     = (float*)(smem + 10240);   // overlay: 16x80 f32
  bf16*  scal_bf = (bf16*) (smem + 15360);   // overlay: 16x64 bf16
  float* spb     = (float*)(smem + 25600);   // overlay: 16x64 f32 (over in0)

  const int e0 = blockIdx.x * 16;
  const int lane = threadIdx.x;
  const int m = lane & 15, h = lane >> 4;
  const int e = e0 + m;

  // speculative prefetch of next tile's streamed inputs (global_prefetch_b8)
  if (e0 + 16 < E_TOT) {
    __builtin_prefetch(latents + (size_t)(e0 + 16 + m) * 128, 0, 3);
    __builtin_prefetch(wigner + (size_t)(e0 + 16 + m) * 64, 0, 3);
    __builtin_prefetch(edge_feat + (size_t)(e0 + 16 + m) * 240 + h * 120, 0, 3);
  }

  // Phase A: latents tile -> LDS bf16; WMMA w = lat@Wlat, we = lat@Wenv
  for (int idx = lane; idx < 16 * 128; idx += 32) {
    int r = idx >> 7, c = idx & 127;
    lat_bf[idx] = (bf16)latents[(size_t)(e0 + r) * 128 + c];
  }
  __syncthreads();
  gemm_bf<4>(lat_bf, 128, WB + OFF_WLAT, wbuf, 224, 224);
  gemm_bf<4>(lat_bf, 128, WB + OFF_WENV, webuf, 112, 112);
  __syncthreads();

  const int ae  = active_edges[e];
  const int nid = edge_index[ae];          // row 0 of edge_index
  const float* wg = wigner + (size_t)e * 64;
  float D1[3][3], D2[5][5];
#pragma unroll
  for (int i = 0; i < 3; ++i)
#pragma unroll
    for (int j = 0; j < 3; ++j) D1[i][j] = wg[i * 8 + j];
#pragma unroll
  for (int i = 0; i < 5; ++i)
#pragma unroll
    for (int j = 0; j < 5; ++j) D2[i][j] = wg[(3 + i) * 8 + (3 + j)];

  // Phase B: gather + rotate + weight; build bf16 A-matrices (h=0 node / h=1 edge)
  {
    const float* f = h ? (edge_feat + (size_t)e * 240) : (node_feat + (size_t)nid * 240);
    const float* wrow = wbuf + m * 224;
    const int sw = h ? 112 : 0, so = h ? 64 : 0;
    for (int c = 0; c < 64; ++c)
      in0_bf[m * 224 + so + c] = (bf16)(f[c] * wrow[sw + c]);
    const int vw = h ? 176 : 64, cb = h ? 32 : 0;
    for (int c = 0; c < 32; ++c) {
      float x0 = f[64 + c * 3], x1 = f[64 + c * 3 + 1], x2 = f[64 + c * 3 + 2];
      float s = wrow[vw + c];
      float r0 = (D1[0][0] * x0 + D1[0][1] * x1 + D1[0][2] * x2) * s;
      float r1 = (D1[1][0] * x0 + D1[1][1] * x1 + D1[1][2] * x2) * s;
      float r2 = (D1[2][0] * x0 + D1[2][1] * x1 + D1[2][2] * x2) * s;
      in0_bf[m * 224 + 128 + cb + c] = (bf16)r1;   // v1 comp 1
      xpn_bf[m * 192 + cb + c]       = (bf16)r2;   // xp  (comp 2)
      xpn_bf[m * 192 + 96 + cb + c]  = (bf16)r0;   // xn  (comp 0)
    }
    const int vw2 = h ? 208 : 96, cb2 = h ? 16 : 0;
    for (int c = 0; c < 16; ++c) {
      float x[5], r[5];
#pragma unroll
      for (int k = 0; k < 5; ++k) x[k] = f[160 + c * 5 + k];
      float s = wrow[vw2 + c];
#pragma unroll
      for (int i = 0; i < 5; ++i) {
        float t = 0.f;
#pragma unroll
        for (int k = 0; k < 5; ++k) t += D2[i][k] * x[k];
        r[i] = t * s;
      }
      in0_bf[m * 224 + 192 + cb2 + c] = (bf16)r[2];
      xpn_bf[m * 192 + 64 + cb2 + c]  = (bf16)r[3];   // xp v2 part
      xpn_bf[m * 192 + 160 + cb2 + c] = (bf16)r[1];   // xn v2 part
      m2_bf [m * 64 + cb2 + c]        = (bf16)r[4];
      m2_bf [m * 64 + 32 + cb2 + c]   = (bf16)r[0];
    }
  }
  __syncthreads();

  // Phase C: main WMMA GEMMs
  gemm_bf<7>(in0_bf, 224, WB + OFF_WM0,  out0, 160, 160);
  gemm_bf<6>(xpn_bf, 192, WB + OFF_WM1C, ypn,   96,  96);
  gemm_bf<2>(m2_bf,   64, WB + OFF_WM2C, y2c,   32,  32);
  __syncthreads();

  // Phase D1: back-rotation + gates (VALU)
  if (h == 0) {
    for (int c = 0; c < 32; ++c) {
      float j0 = ypn[m * 96 + 48 + c];       // yn
      float j1 = out0[m * 160 + 112 + c];    // y1_0
      float j2 = ypn[m * 96 + c];            // yp
      float g = sigm(out0[m * 160 + 64 + c]);
#pragma unroll
      for (int i = 0; i < 3; ++i)
        y1b[m * 96 + c * 3 + i] = (D1[0][i] * j0 + D1[1][i] * j1 + D1[2][i] * j2) * g;
    }
  } else {
    for (int c = 0; c < 16; ++c) {
      float j[5];
      j[0] = y2c[m * 32 + 16 + c];           // yn2
      j[1] = ypn[m * 96 + 48 + 32 + c];      // yn[32+c]
      j[2] = out0[m * 160 + 144 + c];        // y2_0
      j[3] = ypn[m * 96 + 32 + c];           // yp[32+c]
      j[4] = y2c[m * 32 + c];                // yp2
      float g = sigm(out0[m * 160 + 96 + c]);
#pragma unroll
      for (int i = 0; i < 5; ++i) {
        float t = 0.f;
#pragma unroll
        for (int k = 0; k < 5; ++k) t += D2[k][i] * j[k];
        y2b[m * 80 + c * 5 + i] = t * g;
      }
    }
    for (int c = 0; c < 64; ++c) {
      float x = out0[m * 160 + c];
      scal_bf[m * 64 + c] = (bf16)(x * sigm(x));   // silu
    }
  }
  __syncthreads();

  // Phase D2: sp = silu(sc)@Wp0 via WMMA
  gemm_bf<2>(scal_bf, 64, WB + OFF_WP0, spb, 64, 64);
  __syncthreads();

  // Final: Wp1/Wp2 (tiny dots), env weighting, fp32 atomic segment-sum
  float* dst = acc + (size_t)nid * 240;
  if (h == 0) {
    for (int d = 0; d < 64; ++d) {
      float v = (spb[m * 64 + d] + bp0[d]) * webuf[m * 112 + d];
      __hip_atomic_fetch_add(dst + d, v, __ATOMIC_RELAXED, __HIP_MEMORY_SCOPE_AGENT);
    }
    for (int d = 0; d < 32; ++d) {
      float wd = webuf[m * 112 + 64 + d];
#pragma unroll
      for (int i = 0; i < 3; ++i) {
        float s = 0.f;
        for (int c = 0; c < 32; ++c) s += y1b[m * 96 + c * 3 + i] * Wp1[c * 32 + d];
        __hip_atomic_fetch_add(dst + 64 + d * 3 + i, s * wd,
                               __ATOMIC_RELAXED, __HIP_MEMORY_SCOPE_AGENT);
      }
    }
  } else {
    for (int d = 0; d < 16; ++d) {
      float wd = webuf[m * 112 + 96 + d];
#pragma unroll
      for (int i = 0; i < 5; ++i) {
        float s = 0.f;
        for (int c = 0; c < 16; ++c) s += y2b[m * 80 + c * 5 + i] * Wp2[c * 16 + d];
        __hip_atomic_fetch_add(dst + 160 + d * 5 + i, s * wd,
                               __ATOMIC_RELAXED, __HIP_MEMORY_SCOPE_AGENT);
      }
    }
  }
}

// ---- node kernel: one wave = 16 nodes --------------------------------------
__global__ __launch_bounds__(32) void node_kernel(
    const float* __restrict__ node_feat, const float* __restrict__ onehot,
    const float* __restrict__ br0, const bf16* __restrict__ WB,
    const float* __restrict__ acc, float* __restrict__ out) {
  __shared__ char smem[44032];
  float* nf   = (float*)(smem);            // 16x240
  float* oh   = (float*)(smem + 15360);    // 16x96 (col 95 padded 0)
  float* resS = (float*)(smem + 21504);    // 16x64
  float* res1 = (float*)(smem + 25600);    // 3 x 16x32
  float* res2 = (float*)(smem + 31744);    // 5 x 16x16
  float* obuf = (float*)(smem + 36864);    // 16x112

  const int n0 = blockIdx.x * 16;
  const int lane = threadIdx.x;
  for (int idx = lane; idx < 16 * 240; idx += 32) {
    int r = idx / 240, c = idx - r * 240;
    int n = n0 + r;
    nf[idx] = (n < NNODES) ? node_feat[(size_t)n * 240 + c] : 0.f;
  }
  for (int idx = lane; idx < 16 * 96; idx += 32) {
    int r = idx / 96, c = idx - r * 96;
    int n = n0 + r;
    oh[idx] = (n < NNODES && c < 95) ? onehot[(size_t)n * 95 + c] : 0.f;
  }
  __syncthreads();

  gemm_f32A<2>(nf, 240, 0, 1, 64, WB + OFF_WR0, resS, 64, 64);
#pragma unroll
  for (int mm = 0; mm < 3; ++mm)
    gemm_f32A<1>(nf, 240, 64 + mm, 3, 32, WB + OFF_WR1, res1 + mm * 512, 32, 32);
#pragma unroll
  for (int mm = 0; mm < 5; ++mm)
    gemm_f32A<1>(nf, 240, 160 + mm, 5, 16, WB + OFF_WR2P, res2 + mm * 256, 16, 16);
  gemm_f32A<3>(oh, 96, 0, 1, 96, WB + OFF_WOC, obuf, 112, 112);
  __syncthreads();

  const int m = lane & 15, h = lane >> 4;
  const int n = n0 + m;
  if (n < NNODES) {
    const float MSG_SCALE = 0.39528471f;   // (160000/25000)^-0.5
    const float C_OLD = 0.89442719f, C_NEW = 0.44721360f;
    for (int f = h * 120; f < h * 120 + 120; ++f) {
      float resv, t;
      if (f < 64) {
        resv = resS[m * 64 + f] + br0[f];
        t = obuf[m * 112 + f];
      } else if (f < 160) {
        int q = f - 64, d = q / 3, i = q - d * 3;
        resv = res1[i * 512 + m * 32 + d];
        t = obuf[m * 112 + 64 + d];
      } else {
        int q = f - 160, d = q / 5, i = q - d * 5;
        resv = res2[i * 256 + m * 16 + d];
        t = obuf[m * 112 + 96 + d];
      }
      float o = C_NEW * MSG_SCALE * acc[(size_t)n * 240 + f] + C_OLD * resv;
      out[(size_t)n * 240 + f] = o * (1.f + t);
    }
  }
}

extern "C" void kernel_launch(void* const* d_in, const int* in_sizes, int n_in,
                              void* d_out, int out_size, void* d_ws, size_t ws_size,
                              hipStream_t stream) {
  const float* latents    = (const float*)d_in[0];
  const float* node_feat  = (const float*)d_in[1];
  const float* edge_feat  = (const float*)d_in[2];
  const float* onehot     = (const float*)d_in[4];
  const int*   edge_index = (const int*)d_in[5];
  const int*   active     = (const int*)d_in[7];
  const float* wigner     = (const float*)d_in[8];
  const float* Wlat = (const float*)d_in[9];
  const float* Wm0  = (const float*)d_in[10];
  const float* Wm1r = (const float*)d_in[11];
  const float* Wm1i = (const float*)d_in[12];
  const float* Wm2r = (const float*)d_in[13];
  const float* Wm2i = (const float*)d_in[14];
  const float* Wenv = (const float*)d_in[15];
  const float* Wp0  = (const float*)d_in[16];
  const float* bp0  = (const float*)d_in[17];
  const float* Wp1  = (const float*)d_in[18];
  const float* Wp2  = (const float*)d_in[19];
  const float* Wr0  = (const float*)d_in[20];
  const float* br0  = (const float*)d_in[21];
  const float* Wr1  = (const float*)d_in[22];
  const float* Wr2  = (const float*)d_in[23];
  const float* Wo0  = (const float*)d_in[24];
  const float* Wo1  = (const float*)d_in[25];
  const float* Wo2  = (const float*)d_in[26];

  float* acc = (float*)d_ws;
  bf16*  WB  = (bf16*)((char*)d_ws + (size_t)ACC_N * 4);

  setup_kernel<<<2048, 256, 0, stream>>>(acc, WB, Wlat, Wm0, Wm1r, Wm1i, Wm2r,
                                         Wm2i, Wenv, Wp0, Wr0, Wr1, Wr2,
                                         Wo0, Wo1, Wo2);
  edge_kernel<<<E_TOT / 16, 32, 0, stream>>>(latents, node_feat, edge_feat,
      edge_index, active, wigner, Wp1, Wp2, bp0, WB, acc);
  node_kernel<<<(NNODES + 15) / 16, 32, 0, stream>>>(node_feat, onehot, br0,
                                                     WB, acc, (float*)d_out);
}
